// GroupQueryAttention_52871047413936
// MI455X (gfx1250) — matrix operand order repaired and checked by reading, not compiled, not run on hardware
//
#include <hip/hip_runtime.h>

#define SEQ 2048
#define EMBD 2048
#define NH 32
#define NG 8
#define HD 64
#define WIN 128

typedef __attribute__((ext_vector_type(16))) __bf16 v16bf;
typedef __attribute__((ext_vector_type(8)))  float  v8f;
typedef __attribute__((ext_vector_type(4)))  int    v4i;

union BF16Frag { v16bf v; unsigned int u[8]; };

#if __has_builtin(__builtin_amdgcn_global_load_async_to_lds_b128) && \
    __has_builtin(__builtin_amdgcn_s_wait_asynccnt)
#define USE_ASYNC_LDS 1
#else
#define USE_ASYNC_LDS 0
#endif

__device__ __forceinline__ unsigned short f2bf(float f) {
  unsigned int x = __builtin_bit_cast(unsigned int, f);
  x += 0x7fffu + ((x >> 16) & 1u);
  return (unsigned short)(x >> 16);
}

#if USE_ASYNC_LDS
__device__ __forceinline__ void async_copy16(const void* g, void* l) {
  __builtin_amdgcn_global_load_async_to_lds_b128(
      (__attribute__((address_space(1))) v4i*)(void*)g,
      (__attribute__((address_space(3))) v4i*)l, 0, 0);
}
#endif

// ---------------------------------------------------------------- cast f32->bf16
__global__ void cast_f32_bf16(const float* __restrict__ in,
                              unsigned short* __restrict__ out, int n) {
  int i = blockIdx.x * blockDim.x + threadIdx.x;
  if (i < n) out[i] = f2bf(in[i]);
}

// ---------------------------------------------------------------- RoPE + bf16
__global__ void rope_to_bf16(const float* __restrict__ T,
                             const float* __restrict__ sin_t,
                             const float* __restrict__ cos_t,
                             const int* __restrict__ pos,
                             unsigned short* __restrict__ out, int heads) {
  int idx = blockIdx.x * blockDim.x + threadIdx.x;   // SEQ*heads*32 threads
  int d = idx & 31;
  int h = (idx >> 5) % heads;
  int s = idx / (32 * heads);
  if (s >= SEQ) return;
  int p = pos[s];
  float cs = cos_t[p * HD + d];       // tables duplicated: [d] == [d+32]
  float sn = sin_t[p * HD + d];
  int base = s * heads * HD + h * HD;
  float t1 = T[base + d], t2 = T[base + d + 32];
  out[base + d]      = f2bf(t1 * cs - t2 * sn);
  out[base + d + 32] = f2bf(t2 * cs + t1 * sn);
}

// ---------------------------------------------------------------- bf16 WMMA GEMM
// C[M,N] = A[M,K] * B[K,N] + bias[N]; double-buffered LDS, async DMA when available.
__global__ __launch_bounds__(256) void gemm_bf16(
    const unsigned short* __restrict__ A,
    const unsigned short* __restrict__ B,
    const float* __restrict__ bias,
    float* __restrict__ C, int N, int K) {
  __shared__ unsigned int sA[2][128 * 16];  // 2 x (128 x 32 bf16)
  __shared__ unsigned int sB[2][32 * 64];   // 2 x (32 x 128 bf16)
  const int tid  = threadIdx.x;
  const int lane = tid & 31;
  const int wid  = tid >> 5;
  const int wm = wid >> 2;        // 0..1 (64 rows each)
  const int wn = wid & 3;         // 0..3 (32 cols each)
  const int row0 = blockIdx.y * 128;
  const int col0 = blockIdx.x * 128;
  const int l16 = lane & 15;
  const int hi  = lane >> 4;

  v8f zero = {0.f,0.f,0.f,0.f,0.f,0.f,0.f,0.f};
  v8f acc[4][2];
#pragma unroll
  for (int i = 0; i < 4; ++i) { acc[i][0] = zero; acc[i][1] = zero; }

  // compute one 32-deep k-step from LDS buffer `buf`
  auto compute = [&](int buf) {
    BF16Frag bf[2];
    const unsigned short* sBs = (const unsigned short*)sB[buf];
#pragma unroll
    for (int nt = 0; nt < 2; ++nt) {
      int n = wn * 32 + nt * 16 + l16;
#pragma unroll
      for (int j = 0; j < 8; ++j) {
        int kb = 16 * hi + 2 * j;
        unsigned int lo = sBs[kb * 128 + n];
        unsigned int hh = sBs[(kb + 1) * 128 + n];
        bf[nt].u[j] = lo | (hh << 16);
      }
    }
#pragma unroll
    for (int mt = 0; mt < 4; ++mt) {
      BF16Frag af;
      int m = wm * 64 + mt * 16 + l16;
#pragma unroll
      for (int j = 0; j < 8; ++j) {
        int kb = 16 * hi + 2 * j;
        af.u[j] = sA[buf][m * 16 + (kb >> 1)];
      }
#pragma unroll
      for (int nt = 0; nt < 2; ++nt)
        acc[mt][nt] = __builtin_amdgcn_wmma_f32_16x16x32_bf16(
            false, af.v, false, bf[nt].v, (short)0, acc[mt][nt], false, false);
    }
  };

#if USE_ASYNC_LDS
  // ---- async double-buffered pipeline: 4 x b128 DMA per lane per tile
  const char* Ab = (const char*)A;
  const char* Bb = (const char*)B;
  auto fill = [&](int buf, int k0) {
#pragma unroll
    for (int op = 0; op < 2; ++op) {               // A tile: 512 x 16B
      int f = op * 256 + tid;
      int r = f >> 2, c = f & 3;                   // 4 x 16B per 32-elem row
      async_copy16(Ab + ((size_t)(row0 + r) * K + k0) * 2 + c * 16,
                   (char*)sA[buf] + f * 16);
    }
#pragma unroll
    for (int op = 0; op < 2; ++op) {               // B tile: 512 x 16B
      int f = op * 256 + tid;
      int r = f >> 4, c = f & 15;                  // 16 x 16B per 128-elem row
      async_copy16(Bb + ((size_t)(k0 + r) * N + col0) * 2 + c * 16,
                   (char*)sB[buf] + f * 16);
    }
  };
  fill(0, 0);
  int cur = 0;
  for (int k0 = 0; k0 + 32 < K; k0 += 32) {
    fill(cur ^ 1, k0 + 32);                        // prefetch next tile
    __builtin_amdgcn_s_wait_asynccnt(4);           // in-order: prev 4 done
    __syncthreads();
    compute(cur);
    __syncthreads();                               // safe to refill `cur`
    cur ^= 1;
  }
  __builtin_amdgcn_s_wait_asynccnt(0);
  __syncthreads();
  compute(cur);
#else
  // ---- fallback: synchronous staging, single buffer
  const unsigned int* Au = (const unsigned int*)A;
  const unsigned int* Bu = (const unsigned int*)B;
  const int Ku2 = K >> 1;
  const int Nu2 = N >> 1;
  for (int k0 = 0; k0 < K; k0 += 32) {
#pragma unroll
    for (int j = 0; j < 8; ++j) {
      int f = j * 256 + tid;
      sA[0][f] = Au[(row0 + (f >> 4)) * Ku2 + (k0 >> 1) + (f & 15)];
    }
#pragma unroll
    for (int j = 0; j < 8; ++j) {
      int f = j * 256 + tid;
      sB[0][f] = Bu[(k0 + (f >> 6)) * Nu2 + (col0 >> 1) + (f & 63)];
    }
    __syncthreads();
    compute(0);
    __syncthreads();
  }
#endif

#pragma unroll
  for (int mt = 0; mt < 4; ++mt)
#pragma unroll
    for (int nt = 0; nt < 2; ++nt) {
      int n = col0 + wn * 32 + nt * 16 + l16;
      float bv = bias[n];
#pragma unroll
      for (int r = 0; r < 8; ++r) {
        int m = row0 + wm * 64 + mt * 16 + r + 8 * hi;
        C[m * N + n] = acc[mt][nt][r] + bv;
      }
    }
}

// ---------------------------------------------------------------- windowed attention
// One wave per (head, 16-query tile). Flash-style online softmax + sink column.
__global__ __launch_bounds__(256) void attn_swa(
    const unsigned short* __restrict__ Qb,   // [SEQ, NH*HD] bf16 (rope'd)
    const unsigned short* __restrict__ Kb,   // [SEQ, NG*HD] bf16 (rope'd)
    const unsigned short* __restrict__ Vb,   // [SEQ, NG*HD] bf16
    const float* __restrict__ sinks,         // [NH]
    unsigned short* __restrict__ Ob) {       // [SEQ, NH*HD] bf16
  __shared__ unsigned int pbuf[8][16 * 16];  // per-wave 16x32 bf16 P relayout
  const int lane = threadIdx.x & 31;
  const int wid  = threadIdx.x >> 5;
  const int task = blockIdx.x * 8 + wid;
  const int h  = task >> 7;           // SEQ/16 = 128 tiles per head
  const int q0 = (task & 127) << 4;
  const int g  = h >> 2;              // H/G = 4 heads per KV group
  const int l16 = lane & 15;
  const int hi  = lane >> 4;

  // Q A-fragments for K-dim 0..31 and 32..63
  BF16Frag qf[2];
  {
    const unsigned int* Qu = (const unsigned int*)Qb;
    const int base = (q0 + l16) * (NH * HD / 2) + h * (HD / 2);
#pragma unroll
    for (int t = 0; t < 2; ++t)
#pragma unroll
      for (int j = 0; j < 8; ++j) {
        int kb = t * 32 + 16 * hi + 2 * j;
        qf[t].u[j] = Qu[base + (kb >> 1)];
      }
  }

  v8f zero = {0.f,0.f,0.f,0.f,0.f,0.f,0.f,0.f};
  v8f ctx[4];
#pragma unroll
  for (int nt = 0; nt < 4; ++nt) ctx[nt] = zero;
  float m_run[8], l_run[8];
#pragma unroll
  for (int r = 0; r < 8; ++r) { m_run[r] = -INFINITY; l_run[r] = 0.f; }

  int js = q0 - (WIN - 1);
  if (js < 0) js = 0;
  js &= ~31;

  const unsigned int* Ku = (const unsigned int*)Kb;
  unsigned int* pb = pbuf[wid];
  unsigned short* pbs = (unsigned short*)pb;

  for (int jt = js; jt <= q0 + 15; jt += 32) {
    // ---- scores: two 16-key halves, each = 2 chained WMMAs over HD=64
    v8f sv[2];
#pragma unroll
    for (int half = 0; half < 2; ++half) {
      BF16Frag kf0, kf1;
      int key = jt + half * 16 + l16;
      if (key > SEQ - 1) key = SEQ - 1;           // masked anyway
      const int kbase = key * (NG * HD / 2) + g * (HD / 2);
#pragma unroll
      for (int j = 0; j < 8; ++j) {
        int kb = 16 * hi + 2 * j;
        kf0.u[j] = Ku[kbase + (kb >> 1)];
        kf1.u[j] = Ku[kbase + ((kb + 32) >> 1)];
      }
      v8f s = zero;
      s = __builtin_amdgcn_wmma_f32_16x16x32_bf16(false, qf[0].v, false, kf0.v, (short)0, s, false, false);
      s = __builtin_amdgcn_wmma_f32_16x16x32_bf16(false, qf[1].v, false, kf1.v, (short)0, s, false, false);
      sv[half] = s;
    }

    // ---- mask + online softmax (row stats shared across each 16-lane group)
    float p0[8], p1[8], resc[8];
#pragma unroll
    for (int r = 0; r < 8; ++r) {
      const int i  = q0 + r + 8 * hi;
      const int j0 = jt + l16;
      const int j1 = jt + 16 + l16;
      float a = sv[0][r] * 0.125f;                // 1/sqrt(64)
      float b = sv[1][r] * 0.125f;
      if (j0 > i || j0 <= i - WIN) a = -INFINITY; // causal + sliding window
      if (j1 > i || j1 <= i - WIN) b = -INFINITY;
      float t = fmaxf(a, b);
      t = fmaxf(t, __shfl_xor(t, 1));
      t = fmaxf(t, __shfl_xor(t, 2));
      t = fmaxf(t, __shfl_xor(t, 4));
      t = fmaxf(t, __shfl_xor(t, 8));
      float mn   = fmaxf(m_run[r], t);
      float mref = fmaxf(mn, -1e37f);             // avoid (-inf)-(-inf)
      float so = __expf(m_run[r] - mref);
      float e0 = __expf(a - mref);
      float e1 = __expf(b - mref);
      float ts = e0 + e1;
      ts += __shfl_xor(ts, 1);
      ts += __shfl_xor(ts, 2);
      ts += __shfl_xor(ts, 4);
      ts += __shfl_xor(ts, 8);
      l_run[r] = l_run[r] * so + ts;
      m_run[r] = mn;
      p0[r] = e0; p1[r] = e1; resc[r] = so;
    }
#pragma unroll
    for (int nt = 0; nt < 4; ++nt)
#pragma unroll
      for (int r = 0; r < 8; ++r) ctx[nt][r] *= resc[r];

    // ---- relayout P (C-layout) -> A-layout via per-wave LDS
#pragma unroll
    for (int r = 0; r < 8; ++r) {
      const int m = r + 8 * hi;
      pbs[m * 32 + l16]      = f2bf(p0[r]);
      pbs[m * 32 + 16 + l16] = f2bf(p1[r]);
    }
    __builtin_amdgcn_wave_barrier();
    asm volatile("s_wait_dscnt 0" ::: "memory");  // in-wave lockstep: DS in order
    __builtin_amdgcn_wave_barrier();
    BF16Frag pf;
#pragma unroll
    for (int j = 0; j < 8; ++j) {
      int kb = 16 * hi + 2 * j;
      pf.u[j] = pb[l16 * 16 + (kb >> 1)];
    }

    // ---- ctx += P(16x32) @ V(32x64): 4 WMMAs over the 4 dim-tiles
#pragma unroll
    for (int nt = 0; nt < 4; ++nt) {
      BF16Frag vf;
      const int dim = nt * 16 + l16;
#pragma unroll
      for (int j = 0; j < 8; ++j) {
        int k0i = jt + 16 * hi + 2 * j;
        int k1i = k0i + 1;
        if (k0i > SEQ - 1) k0i = SEQ - 1;
        if (k1i > SEQ - 1) k1i = SEQ - 1;
        unsigned int lo = Vb[k0i * (NG * HD) + g * HD + dim];
        unsigned int hh = Vb[k1i * (NG * HD) + g * HD + dim];
        vf.u[j] = lo | (hh << 16);
      }
      ctx[nt] = __builtin_amdgcn_wmma_f32_16x16x32_bf16(
          false, pf.v, false, vf.v, (short)0, ctx[nt], false, false);
    }
  }

  // ---- fold in attention sink, normalize, write bf16 ctx
  const float snk = sinks[h];
  float outsc[8];
#pragma unroll
  for (int r = 0; r < 8; ++r) {
    float mn = fmaxf(m_run[r], snk);
    float denom = l_run[r] * __expf(m_run[r] - mn) + __expf(snk - mn);
    outsc[r] = __expf(m_run[r] - mn) / denom;
  }
#pragma unroll
  for (int nt = 0; nt < 4; ++nt)
#pragma unroll
    for (int r = 0; r < 8; ++r) {
      int row = q0 + r + 8 * hi;
      Ob[row * (NH * HD) + h * HD + nt * 16 + l16] = f2bf(ctx[nt][r] * outsc[r]);
    }
}

// ---------------------------------------------------------------- launcher
extern "C" void kernel_launch(void* const* d_in, const int* in_sizes, int n_in,
                              void* d_out, int out_size, void* d_ws, size_t ws_size,
                              hipStream_t stream) {
  (void)in_sizes; (void)n_in; (void)out_size; (void)ws_size;
  const float* x     = (const float*)d_in[0];
  const int*   pos   = (const int*)d_in[1];
  // d_in[2] attn_mask: implied by causal+window logic, unused
  const float* sin_t = (const float*)d_in[3];
  const float* cos_t = (const float*)d_in[4];
  const float* Wq = (const float*)d_in[5];
  const float* bq = (const float*)d_in[6];
  const float* Wk = (const float*)d_in[7];
  const float* bk = (const float*)d_in[8];
  const float* Wv = (const float*)d_in[9];
  const float* bv = (const float*)d_in[10];
  const float* Wo = (const float*)d_in[11];
  const float* bo = (const float*)d_in[12];
  const float* sk = (const float*)d_in[13];
  float* out = (float*)d_out;

  char* p = (char*)d_ws;
  auto take = [&](size_t bytes) {
    char* r = p; p += (bytes + 255) & ~(size_t)255; return r;
  };
  unsigned short* xb  = (unsigned short*)take((size_t)SEQ * EMBD * 2);
  unsigned short* Wqb = (unsigned short*)take((size_t)EMBD * EMBD * 2);
  unsigned short* Wkb = (unsigned short*)take((size_t)EMBD * NG * HD * 2);
  unsigned short* Wvb = (unsigned short*)take((size_t)EMBD * NG * HD * 2);
  unsigned short* Wob = (unsigned short*)take((size_t)EMBD * EMBD * 2);
  float* Qf = (float*)take((size_t)SEQ * NH * HD * 4);
  float* Kf = (float*)take((size_t)SEQ * NG * HD * 4);
  float* Vf = (float*)take((size_t)SEQ * NG * HD * 4);
  unsigned short* Qbf = (unsigned short*)take((size_t)SEQ * NH * HD * 2);
  unsigned short* Kbf = (unsigned short*)take((size_t)SEQ * NG * HD * 2);
  unsigned short* Vbf = (unsigned short*)take((size_t)SEQ * NG * HD * 2);
  unsigned short* Cb  = (unsigned short*)take((size_t)SEQ * NH * HD * 2);

  auto cast = [&](const float* in, unsigned short* o, int n) {
    cast_f32_bf16<<<(n + 255) / 256, 256, 0, stream>>>(in, o, n);
  };
  cast(x,  xb,  SEQ * EMBD);
  cast(Wq, Wqb, EMBD * EMBD);
  cast(Wk, Wkb, EMBD * NG * HD);
  cast(Wv, Wvb, EMBD * NG * HD);
  cast(Wo, Wob, EMBD * EMBD);

  // QKV projections (fp32 accumulate)
  gemm_bf16<<<dim3(EMBD / 128, SEQ / 128), 256, 0, stream>>>(xb, Wqb, bq, Qf, EMBD, EMBD);
  gemm_bf16<<<dim3(NG * HD / 128, SEQ / 128), 256, 0, stream>>>(xb, Wkb, bk, Kf, NG * HD, EMBD);
  gemm_bf16<<<dim3(NG * HD / 128, SEQ / 128), 256, 0, stream>>>(xb, Wvb, bv, Vf, NG * HD, EMBD);

  // RoPE + bf16
  rope_to_bf16<<<(SEQ * NH * 32) / 256, 256, 0, stream>>>(Qf, sin_t, cos_t, pos, Qbf, NH);
  rope_to_bf16<<<(SEQ * NG * 32) / 256, 256, 0, stream>>>(Kf, sin_t, cos_t, pos, Kbf, NG);
  cast(Vf, Vbf, SEQ * NG * HD);

  // windowed flash attention with sinks -> bf16 ctx
  attn_swa<<<(NH * (SEQ / 16)) / 8, 256, 0, stream>>>(Qbf, Kbf, Vbf, sk, Cb);

  // output projection -> fp32 out
  gemm_bf16<<<dim3(EMBD / 128, SEQ / 128), 256, 0, stream>>>(Cb, Wob, bo, out, EMBD, EMBD);
}
